// MGModel_87351044866594
// MI455X (gfx1250) — matrix operand order repaired and checked by reading, not compiled, not run on hardware
//
#include <hip/hip_runtime.h>

typedef __attribute__((ext_vector_type(16))) _Float16 v16h;
typedef __attribute__((ext_vector_type(8)))  float    v8f;

static constexpr int D    = 64;
static constexpr int OUTF = 16;
#define BN_EPS 1e-5f

// ---------------------------------------------------------------------------
// Elementwise / utility kernels
// ---------------------------------------------------------------------------
__global__ void k_zero(float* __restrict__ p, long long n) {
  long long t = (long long)blockIdx.x * blockDim.x + threadIdx.x;
  if (t < n) p[t] = 0.0f;
}

// Convert a 64x64 f32 weight (row-major, K-major) into f16 *fragment-major*
// layout: for frag = kb*4 + nc (kb in {0,1}, nc in {0..3}), lane in {0..31},
// the 16 halves lane needs for WMMA B are contiguous:
//   dst[((kb*4+nc)*32 + lane)*16 + i] = W[(kb*32 + (i<8 ? 8*(lane>>4) + i : 16 + 8*(lane>>4) + i-8))*64 + nc*16 + (lane&15)]
__global__ void k_w_swizzle_f16(const float* __restrict__ W, _Float16* __restrict__ d, int n) {
  int t = blockIdx.x * blockDim.x + threadIdx.x;
  if (t >= n) return;
  int i    = t & 15;
  int lane = (t >> 4) & 31;
  int frag = t >> 9;  // 0..7
  int nc   = frag & 3;
  int kb   = frag >> 2;
  int g    = lane >> 4;
  int col  = nc * 16 + (lane & 15);
  int k    = kb * 32 + ((i < 8) ? (8 * g + i) : (16 + 8 * g + (i - 8)));
  d[t] = (_Float16)W[k * D + col];
}

__global__ void k_degree(const int* __restrict__ dst, float* __restrict__ cnt, int E) {
  int t = blockIdx.x * blockDim.x + threadIdx.x;
  if (t < E) atomicAdd(&cnt[dst[t]], 1.0f);
}

// one thread per (edge, 4-float group): b128 gather + 4 atomic f32 adds
__global__ void k_scatter_add(const float* __restrict__ feat, const int* __restrict__ src,
                              const int* __restrict__ dst, float* __restrict__ out,
                              long long E16) {
  long long t = (long long)blockIdx.x * blockDim.x + threadIdx.x;
  if (t >= E16) return;
  int e = (int)(t >> 4);
  int q = ((int)t & 15) << 2;
  int s  = src[e];
  int d2 = dst[e];
  const float4 v = *(const float4*)(feat + (size_t)s * D + q);
  float* o = out + (size_t)d2 * D + q;
  atomicAdd(o + 0, v.x);
  atomicAdd(o + 1, v.y);
  atomicAdd(o + 2, v.z);
  atomicAdd(o + 3, v.w);
}

__global__ void k_mean_div(float* __restrict__ agg, const float* __restrict__ cnt, long long n) {
  long long t = (long long)blockIdx.x * blockDim.x + threadIdx.x;
  if (t < n) agg[t] /= fmaxf(cnt[t >> 6], 1.0f);
}

// zero rows belonging to isolated nodes (cnt==0): layer-0 masked-bias semantics
__global__ void k_mask_iso(float* __restrict__ h, const float* __restrict__ cnt, long long n) {
  long long t = (long long)blockIdx.x * blockDim.x + threadIdx.x;
  if (t < n && !(cnt[t >> 6] > 0.0f)) h[t] = 0.0f;
}

__global__ void k_add_inplace(float* __restrict__ a, const float* __restrict__ b, long long n) {
  long long t = (long long)blockIdx.x * blockDim.x + threadIdx.x;
  if (t < n) a[t] += b[t];
}

// ---------------------------------------------------------------------------
// BatchNorm (training-mode, biased var) + ELU
// ---------------------------------------------------------------------------
__global__ void k_bn_stats(const float* __restrict__ h, float* __restrict__ stats, int Nrows) {
  __shared__ float ssum[D];
  __shared__ float ssq[D];
  int c  = threadIdx.x & 63;
  int rg = threadIdx.x >> 6;  // 0..3
  if (rg == 0) { ssum[c] = 0.0f; ssq[c] = 0.0f; }
  __syncthreads();
  float s = 0.0f, q = 0.0f;
  int base = blockIdx.x * 256;
#pragma unroll 4
  for (int i = 0; i < 64; ++i) {
    int row = base + i * 4 + rg;
    if (row < Nrows) {
      float v = h[(size_t)row * D + c];
      s += v;
      q += v * v;
    }
  }
  atomicAdd(&ssum[c], s);
  atomicAdd(&ssq[c], q);
  __syncthreads();
  if (rg == 0) {
    atomicAdd(&stats[c], ssum[c]);
    atomicAdd(&stats[D + c], ssq[c]);
  }
}

__global__ void k_bn_elu(float* __restrict__ h, const float* __restrict__ stats,
                         const float* __restrict__ gamma, const float* __restrict__ beta,
                         long long n, float invN) {
  long long t = (long long)blockIdx.x * blockDim.x + threadIdx.x;
  if (t >= n) return;
  int c = (int)t & 63;
  float mu  = stats[c] * invN;
  float var = stats[D + c] * invN - mu * mu;
  float v = (h[t] - mu) * rsqrtf(var + BN_EPS) * gamma[c] + beta[c];
  h[t] = (v > 0.0f) ? v : (expf(v) - 1.0f);
}

// ---------------------------------------------------------------------------
// WMMA GEMM:  Y[N,64] = A@W (+ A2@W2) + bias (+ optional ReLU)
// One wave per 16-row tile; f16 A/B fragments, f32 accumulate.
// A fragments: CDNA5 ISA 7.12.2 16-bit A 16x32 layout, converted inline from f32.
// B fragments: single contiguous v16h load from pre-swizzled weights.
// All control flow is compile-time (template) or wave-uniform: EXEC stays
// all-ones through every v_wmma.
// ---------------------------------------------------------------------------
__device__ __forceinline__ v16h load_a_frag(const float* __restrict__ A, int row0, int kblock,
                                            int lane, int Nrows) {
  int g = (lane >> 4) & 1;
  int r = row0 + (lane & 15);
  if (r >= Nrows) r = Nrows - 1;
  const float* p = A + (size_t)r * D + kblock + 8 * g;
  v16h a;
#pragma unroll
  for (int i = 0; i < 8; ++i) a[i] = (_Float16)p[i];          // K = 8g + i
#pragma unroll
  for (int i = 0; i < 8; ++i) a[8 + i] = (_Float16)p[16 + i]; // K = 16 + 8g + i
  return a;
}

__device__ __forceinline__ v16h load_b_frag(const _Float16* __restrict__ Wsw, int kb, int nc,
                                            int lane) {
  return *(const v16h*)(Wsw + (((size_t)(kb * 4 + nc) * 32 + lane) << 4));
}

template <bool DUAL, bool RELU>
__global__ void k_gemm64(const float* __restrict__ A, const _Float16* __restrict__ W,
                         const float* __restrict__ A2, const _Float16* __restrict__ W2,
                         const float* __restrict__ bias, float* __restrict__ Y, int Nrows) {
  int lane = threadIdx.x & 31;
  int wave = threadIdx.x >> 5;
  int tile = blockIdx.x * 8 + wave;
  int numTiles = (Nrows + 15) >> 4;
  if (tile >= numTiles) return;  // wave-uniform
  int row0 = tile << 4;
  bool fullTile = (row0 + 16) <= Nrows;  // wave-uniform

  v16h a0 = load_a_frag(A, row0, 0, lane, Nrows);
  v16h a1 = load_a_frag(A, row0, 32, lane, Nrows);
  v16h a20 = {}, a21 = {};
  if (DUAL) {
    a20 = load_a_frag(A2, row0, 0, lane, Nrows);
    a21 = load_a_frag(A2, row0, 32, lane, Nrows);
  }
  int g  = lane >> 4;
  int nl = lane & 15;
#pragma unroll
  for (int nc = 0; nc < 4; ++nc) {
    int col = nc * 16 + nl;
    v16h b0 = load_b_frag(W, 0, nc, lane);
    v16h b1 = load_b_frag(W, 1, nc, lane);
    v8f c = {};
    c = __builtin_amdgcn_wmma_f32_16x16x32_f16(false, a0, false, b0, (short)0, c, false, false);
    c = __builtin_amdgcn_wmma_f32_16x16x32_f16(false, a1, false, b1, (short)0, c, false, false);
    if (DUAL) {
      v16h d0 = load_b_frag(W2, 0, nc, lane);
      v16h d1 = load_b_frag(W2, 1, nc, lane);
      c = __builtin_amdgcn_wmma_f32_16x16x32_f16(false, a20, false, d0, (short)0, c, false, false);
      c = __builtin_amdgcn_wmma_f32_16x16x32_f16(false, a21, false, d1, (short)0, c, false, false);
    }
    float bcol = bias[col];  // one load per column tile, hoisted out of r-loop
    float* __restrict__ yp = Y + (size_t)(row0 + 8 * g) * D + col;
    if (fullTile) {
#pragma unroll
      for (int r = 0; r < 8; ++r) {
        float v = c[r] + bcol;
        if (RELU) v = fmaxf(v, 0.0f);
        yp[(size_t)r * D] = v;
      }
    } else {
#pragma unroll
      for (int r = 0; r < 8; ++r) {
        int row = row0 + r + 8 * g;
        if (row < Nrows) {
          float v = c[r] + bcol;
          if (RELU) v = fmaxf(v, 0.0f);
          Y[(size_t)row * D + col] = v;
        }
      }
    }
  }
}

// ---------------------------------------------------------------------------
// Global mean pool + linear head
// ---------------------------------------------------------------------------
__global__ void k_pool(const float* __restrict__ h, const int* __restrict__ batch,
                       float* __restrict__ pooled, float* __restrict__ pcnt, long long n) {
  long long t = (long long)blockIdx.x * blockDim.x + threadIdx.x;
  if (t >= n) return;
  int node = (int)(t >> 6);
  int c    = (int)t & 63;
  int g    = batch[node];
  atomicAdd(&pooled[(size_t)g * D + c], h[t]);
  if (c == 0) atomicAdd(&pcnt[g], 1.0f);
}

__global__ void k_head(const float* __restrict__ pooled, const float* __restrict__ pcnt,
                       const float* __restrict__ Wout, const float* __restrict__ bout,
                       float* __restrict__ out, int G) {
  int t = blockIdx.x * blockDim.x + threadIdx.x;
  if (t >= G * OUTF) return;
  int g = t >> 4;
  int o = t & 15;
  float inv = 1.0f / fmaxf(pcnt[g], 1.0f);
  float acc = bout[o];
#pragma unroll 8
  for (int k = 0; k < D; ++k) acc += pooled[(size_t)g * D + k] * inv * Wout[k * OUTF + o];
  out[t] = acc;
}

// ---------------------------------------------------------------------------
// Launcher
// ---------------------------------------------------------------------------
extern "C" void kernel_launch(void* const* d_in, const int* in_sizes, int n_in,
                              void* d_out, int out_size, void* d_ws, size_t ws_size,
                              hipStream_t stream) {
  (void)n_in; (void)ws_size;
  const float* x      = (const float*)d_in[0];
  const int*   ei     = (const int*)d_in[1];
  const int*   batch  = (const int*)d_in[2];
  const float* W0     = (const float*)d_in[3];
  const float* b0     = (const float*)d_in[4];
  const float* Wl     = (const float*)d_in[5];
  const float* bl     = (const float*)d_in[6];
  const float* Wr     = (const float*)d_in[7];
  const float* W2a    = (const float*)d_in[8];
  const float* b2a    = (const float*)d_in[9];
  const float* W2b    = (const float*)d_in[10];
  const float* b2b    = (const float*)d_in[11];
  const float* Wrel   = (const float*)d_in[12];
  const float* brel   = (const float*)d_in[13];
  const float* Wroot  = (const float*)d_in[14];
  const float* gammas = (const float*)d_in[15];
  const float* betas  = (const float*)d_in[16];
  const float* Wout   = (const float*)d_in[17];
  const float* bout   = (const float*)d_in[18];
  float* out = (float*)d_out;

  const int N = in_sizes[0] / D;
  const int E = in_sizes[1] / 2;
  const int G = out_size / OUTF;
  const int* src = ei;
  const int* dst = ei + E;

  // workspace carve-out
  char* wsp = (char*)d_ws;
  auto carve = [&](size_t bytes) {
    char* p = wsp;
    wsp += (bytes + 255) & ~(size_t)255;
    return p;
  };
  float*    hA     = (float*)carve((size_t)N * D * 4);
  float*    hB     = (float*)carve((size_t)N * D * 4);
  float*    hC     = (float*)carve((size_t)N * D * 4);
  float*    cnt    = (float*)carve((size_t)N * 4);
  float*    stats  = (float*)carve((size_t)2 * D * 4);
  float*    pooled = (float*)carve((size_t)G * D * 4);
  float*    pcnt   = (float*)carve((size_t)G * 4);
  _Float16* W16    = (_Float16*)carve((size_t)7 * D * D * 2);

  const long long ND  = (long long)N * D;
  const long long E16 = (long long)E * 16;
  const int TB = 256;
  auto nb = [](long long n, int tb) { return (unsigned)((n + tb - 1) / tb); };

  const unsigned ndGrid   = nb(ND, TB);
  const unsigned scatGrid = nb(E16, TB);
  const unsigned gemmGrid = (unsigned)((((N + 15) / 16) + 7) / 8);
  const float invN = 1.0f / (float)N;

  // convert + swizzle the seven 64x64 weight matrices to fragment-major f16
  const float* Wsrc[7] = {W0, Wl, Wr, W2a, W2b, Wrel, Wroot};
  for (int i = 0; i < 7; ++i)
    k_w_swizzle_f16<<<nb(D * D, TB), TB, 0, stream>>>(Wsrc[i], W16 + (size_t)i * D * D, D * D);
  _Float16* fW0    = W16 + 0 * D * D;
  _Float16* fWl    = W16 + 1 * D * D;
  _Float16* fWr    = W16 + 2 * D * D;
  _Float16* fW2a   = W16 + 3 * D * D;
  _Float16* fW2b   = W16 + 4 * D * D;
  _Float16* fWrel  = W16 + 5 * D * D;
  _Float16* fWroot = W16 + 6 * D * D;

  // in-degree (shared by both mean layers; also the isolated-node mask)
  k_zero<<<nb(N, TB), TB, 0, stream>>>(cnt, N);
  k_degree<<<nb(E, TB), TB, 0, stream>>>(dst, cnt, E);

  auto bn = [&](float* h, int layer) {
    k_zero<<<1, 2 * D, 0, stream>>>(stats, 2 * D);
    k_bn_stats<<<nb(N, 256), 256, 0, stream>>>(h, stats, N);
    k_bn_elu<<<ndGrid, TB, 0, stream>>>(h, stats, gammas + layer * D, betas + layer * D, ND, invN);
  };

  // ---- layer 0: custom_mp (mean agg -> linear; bias vanishes on isolated) ----
  k_zero<<<ndGrid, TB, 0, stream>>>(hC, ND);
  k_scatter_add<<<scatGrid, TB, 0, stream>>>(x, src, dst, hC, E16);
  k_mean_div<<<ndGrid, TB, 0, stream>>>(hC, cnt, ND);
  k_gemm64<false, false><<<gemmGrid, TB, 0, stream>>>(hC, fW0, nullptr, nullptr, b0, hA, N);
  k_mask_iso<<<ndGrid, TB, 0, stream>>>(hA, cnt, ND);  // rows with deg==0 -> 0
  bn(hA, 0);

  // ---- layer 1: SAGE (agg@Wl + h@Wr + bl), fused dual-GEMM ----
  k_zero<<<ndGrid, TB, 0, stream>>>(hC, ND);
  k_scatter_add<<<scatGrid, TB, 0, stream>>>(hA, src, dst, hC, E16);
  k_mean_div<<<ndGrid, TB, 0, stream>>>(hC, cnt, ND);
  k_gemm64<true, false><<<gemmGrid, TB, 0, stream>>>(hC, fWl, hA, fWr, bl, hB, N);
  bn(hB, 1);

  // ---- layer 2: GIN  relu((h + sum)@W2a + b2a) @ W2b + b2b ----
  k_zero<<<ndGrid, TB, 0, stream>>>(hC, ND);
  k_scatter_add<<<scatGrid, TB, 0, stream>>>(hB, src, dst, hC, E16);
  k_add_inplace<<<ndGrid, TB, 0, stream>>>(hC, hB, ND);
  k_gemm64<false, true><<<gemmGrid, TB, 0, stream>>>(hC, fW2a, nullptr, nullptr, b2a, hA, N);
  k_gemm64<false, false><<<gemmGrid, TB, 0, stream>>>(hA, fW2b, nullptr, nullptr, b2b, hB, N);
  bn(hB, 2);

  // ---- layer 3: GraphConv (sum@Wrel + brel + h@Wroot), fused dual-GEMM ----
  k_zero<<<ndGrid, TB, 0, stream>>>(hC, ND);
  k_scatter_add<<<scatGrid, TB, 0, stream>>>(hB, src, dst, hC, E16);
  k_gemm64<true, false><<<gemmGrid, TB, 0, stream>>>(hC, fWrel, hB, fWroot, brel, hA, N);
  bn(hA, 3);

  // ---- global mean pool + head ----
  k_zero<<<nb((long long)G * D, TB), TB, 0, stream>>>(pooled, (long long)G * D);
  k_zero<<<nb(G, TB), TB, 0, stream>>>(pcnt, G);
  k_pool<<<ndGrid, TB, 0, stream>>>(hA, batch, pooled, pcnt, ND);
  k_head<<<nb(G * OUTF, TB), TB, 0, stream>>>(pooled, pcnt, Wout, bout, out, G);
}
